// LearnableGating_51994874085984
// MI455X (gfx1250) — matrix operand order repaired and checked by reading, not compiled, not run on hardware
//
#include <hip/hip_runtime.h>
#include <hip/hip_bf16.h>
#include <cmath>

// ---------------------------------------------------------------------------
// MI455X / gfx1250, wave32. All heavy math via v_wmma_f32_16x16x32_bf16.
//   B=4096, M=3, K(experts)=4, E=1024, H=4096, rows R = B*M = 12288
//
// Weights are pre-swizzled (fp32 -> bf16) into WMMA B-fragment tile order so
// the GEMM hot loops fetch each 32x16 B tile with two coalesced
// global_load_b128 per lane. The fused expert kernel uses 32-row tiles with
// H processed in two 2048-wide chunks (down-proj partial accumulation) so the
// hidden activations stay in a 128 KB LDS tile while halving the dominant
// L2 weight traffic vs 16-row tiles.
// ---------------------------------------------------------------------------

#define BB   4096
#define MM   3
#define KEXP 4
#define EE   1024
#define HH   4096
#define RTOT (BB * MM)

#define ROWS  32              // rows per expert block
#define CH    2048            // H chunk width
#define NCH   (HH / CH)       // 2 chunks

typedef __attribute__((ext_vector_type(16))) __bf16 v16bf;
typedef __attribute__((ext_vector_type(8)))  __bf16 v8bf;
typedef __attribute__((ext_vector_type(8)))  float  v8f;

// LDS strides (elements), padded so row-to-row skew = 4 dwords -> 16 rows hit
// 16 distinct 4-bank groups => conflict-free ds_load_b128 fragment fetches.
#define XSTR  (EE + 8)        // 1032 bf16 (2064 B, 16B aligned)
#define CSTR  (CH + 8)        // 2056 bf16 (4112 B, 16B aligned)
#define GSTR3 (3 * EE + 8)    // 3080 bf16 (6160 B, 16B aligned)
#define RSTR  (EE + 4)        // 1028 f32  (skew 4 dwords)

__device__ __forceinline__ float geluf(float x) {
    return 0.5f * x * (1.0f + erff(x * 0.70710678118654752440f));
}

__device__ __forceinline__ v8f wmma_bf16(v16bf a, v16bf b, v8f c) {
    // D = A(16x32 bf16) * B(32x16 bf16) + C(16x16 f32)
    return __builtin_amdgcn_wmma_f32_16x16x32_bf16(
        /*neg_a=*/false, a, /*neg_b=*/false, b,
        /*c_mod=*/(short)0, c, /*reuse_a=*/false, /*reuse_b=*/false);
}

// A fragment (16x32 bf16) from an LDS row. lane m=l%16, half=l/16;
// element e: k = half*16 + e (matches swizzled B). Two b128 loads cover it.
__device__ __forceinline__ v16bf afrag(const __bf16* rowk, int half) {
    v8bf lo = *(const v8bf*)(rowk + half * 16);      // k = half*16 .. +7
    v8bf hi = *(const v8bf*)(rowk + half * 16 + 8);  // k = half*16+8 .. +7
    return __builtin_shufflevector(lo, hi, 0, 1, 2, 3, 4, 5, 6, 7,
                                   8, 9, 10, 11, 12, 13, 14, 15);
}

// B fragment (32x16 bf16) from a pre-swizzled weight tile: lane l's 16 halves
// are contiguous at tile*512 + l*16 -> two coalesced global_load_b128.
__device__ __forceinline__ v16bf bfrag_sw(const __bf16* __restrict__ base,
                                          int tile, int lane) {
    const __bf16* p = base + (size_t)tile * 512 + lane * 16;
    v8bf lo = *(const v8bf*)p;
    v8bf hi = *(const v8bf*)(p + 8);
    return __builtin_shufflevector(lo, hi, 0, 1, 2, 3, 4, 5, 6, 7,
                                   8, 9, 10, 11, 12, 13, 14, 15);
}

// ---------------------------------------------------------------------------
// Kernel 0: swizzle a (Kdim x N) fp32 row-major weight into bf16 B-fragment
// tile order. tile = kt*(N/16)+nt holds the 32x16 K x N subblock; within it,
// lane l = 16*half + n stores elements e=0..15 <-> K = kt*32 + half*16 + e,
// column = nt*16 + n. blockIdx.y indexes stacked expert matrices.
// ---------------------------------------------------------------------------
__global__ void wswizzle_kernel(const float* __restrict__ src,
                                __bf16* __restrict__ dst,
                                int Kdim, int N) {
    const size_t mat = (size_t)Kdim * N;
    src += (size_t)blockIdx.y * mat;
    dst += (size_t)blockIdx.y * mat;
    const int gid = blockIdx.x * blockDim.x + threadIdx.x;
    if (gid >= (int)(mat >> 4)) return;
    const int tile = gid >> 5;
    const int lane = gid & 31;
    const int ntl  = N >> 4;
    const int kt = tile / ntl, nt = tile - kt * ntl;
    const int half = lane >> 4, n = lane & 15;
    const float* s = src + (size_t)(kt * 32 + half * 16) * N + nt * 16 + n;
    v8bf lo, hi;
#pragma unroll
    for (int e = 0; e < 8; ++e) lo[e] = (__bf16)s[(size_t)e * N];
#pragma unroll
    for (int e = 0; e < 8; ++e) hi[e] = (__bf16)s[(size_t)(e + 8) * N];
    __bf16* d = dst + (size_t)tile * 512 + lane * 16;
    *(v8bf*)d = lo;
    *(v8bf*)(d + 8) = hi;
}

// ---------------------------------------------------------------------------
// Kernel 1: g = LayerNorm(x_flat @ Wg1 + bg1) -> GELU   (4096 x 1024, K=3072)
// 16 rows/block, 8 waves, 128 cols/wave. Dynamic LDS: bf16 A tile + f32 row buf.
// ---------------------------------------------------------------------------
__global__ void gate1_kernel(const float* __restrict__ x,
                             const __bf16* __restrict__ Wg1s,
                             const float* __restrict__ bg1,
                             const float* __restrict__ gamma,
                             const float* __restrict__ beta,
                             float* __restrict__ gact) {
    extern __shared__ __align__(16) unsigned char smem[];
    __bf16* sA = (__bf16*)smem;                       // 16 x GSTR3
    float*  sR = (float*)(smem + 16 * GSTR3 * 2);     // 16 x RSTR

    const int tid  = threadIdx.x;
    const int wave = tid >> 5, lane = tid & 31;
    const int half = lane >> 4, n = lane & 15;
    const int rowBase = blockIdx.x * 16;

    // stage A tile (16 x 3072) fp32 -> bf16
    for (int idx = tid; idx < 16 * 3 * EE; idx += 256) {
        int r = idx / (3 * EE), c = idx % (3 * EE);
        sA[r * GSTR3 + c] = (__bf16)x[(size_t)(rowBase + r) * (3 * EE) + c];
    }
    __syncthreads();

    v8f cAcc[8];
#pragma unroll
    for (int t = 0; t < 8; ++t) cAcc[t] = (v8f){0, 0, 0, 0, 0, 0, 0, 0};

    for (int kk = 0; kk < (3 * EE) / 32; ++kk) {               // 96 k-steps
        v16bf a = afrag(&sA[(lane & 15) * GSTR3 + kk * 32], half);
#pragma unroll
        for (int t = 0; t < 8; ++t) {
            v16bf b = bfrag_sw(Wg1s, kk * (EE / 16) + wave * 8 + t, lane);
            cAcc[t] = wmma_bf16(a, b, cAcc[t]);
        }
    }

    // spill pre-LN values (+bias) to LDS for the row reductions
#pragma unroll
    for (int t = 0; t < 8; ++t) {
        int col  = wave * 128 + t * 16 + n;
        float bv = bg1[col];
#pragma unroll
        for (int v = 0; v < 8; ++v)
            sR[(v + half * 8) * RSTR + col] = cAcc[t][v] + bv;
    }
    __syncthreads();

    // LayerNorm + GELU: wave w handles rows 2w, 2w+1 (butterfly reductions)
#pragma unroll
    for (int rr = 0; rr < 2; ++rr) {
        int row = 2 * wave + rr;
        float s = 0.f, s2 = 0.f;
#pragma unroll 8
        for (int j = 0; j < EE / 32; ++j) {
            float v = sR[row * RSTR + lane + 32 * j];
            s += v; s2 += v * v;
        }
#pragma unroll
        for (int off = 16; off >= 1; off >>= 1) {
            s  += __shfl_xor(s, off, 32);
            s2 += __shfl_xor(s2, off, 32);
        }
        float mean = s * (1.0f / EE);
        float var  = s2 * (1.0f / EE) - mean * mean;
        float rstd = rsqrtf(var + 1e-5f);
#pragma unroll 8
        for (int j = 0; j < EE / 32; ++j) {
            int c = lane + 32 * j;
            float v = sR[row * RSTR + c];
            float y = (v - mean) * rstd * gamma[c] + beta[c];
            gact[(size_t)(rowBase + row) * EE + c] = geluf(y);
        }
    }
}

// ---------------------------------------------------------------------------
// Kernel 2: logits = gact @ Wg2 + bg2; softmax over 12 -> gates (B x 12)
// ---------------------------------------------------------------------------
__global__ void gate2_kernel(const float* __restrict__ gact,
                             const float* __restrict__ Wg2,
                             const float* __restrict__ bg2,
                             float* __restrict__ gates) {
    __shared__ float red[12][64];
    __shared__ float lg[12];
    const int tid = threadIdx.x;
    const int b   = blockIdx.x;

    float acc[12];
#pragma unroll
    for (int j = 0; j < 12; ++j) acc[j] = 0.f;
    for (int c = tid; c < EE; c += 64) {
        float g = gact[(size_t)b * EE + c];
#pragma unroll
        for (int j = 0; j < 12; ++j) acc[j] += g * Wg2[c * 12 + j];
    }
#pragma unroll
    for (int j = 0; j < 12; ++j) red[j][tid] = acc[j];
    __syncthreads();
    if (tid < 12) {
        float s = 0.f;
        for (int i = 0; i < 64; ++i) s += red[tid][i];
        lg[tid] = s + bg2[tid];
    }
    __syncthreads();
    if (tid == 0) {
        float mx = lg[0];
        for (int j = 1; j < 12; ++j) mx = fmaxf(mx, lg[j]);
        float den = 0.f;
        float e[12];
        for (int j = 0; j < 12; ++j) { e[j] = expf(lg[j] - mx); den += e[j]; }
        float inv = 1.0f / den;
        for (int j = 0; j < 12; ++j) gates[(size_t)b * 12 + j] = e[j] * inv;
    }
}

// ---------------------------------------------------------------------------
// Kernel 3: fused experts. One block = 32 (b,m) rows, 512 thr (16 waves).
// Waves 0-7 own rows 0-15, waves 8-15 own rows 16-31. Per expert k and per
// H-chunk c (2 x 2048):
//   phase A: Hc[16x2048/rowgrp] = gelu(X @ W1[k][:,chunk] + b1) -> bf16 LDS
//   phase B: cAcc += Hc @ W2[k][chunk,:]   (partial down-proj accumulation)
// After both chunks: outAcc += gate * (cAcc + b2). Output written once.
// 384 blocks; weight traffic halved vs 16-row tiles.
// ---------------------------------------------------------------------------
__global__ void __launch_bounds__(512)
expert_kernel(const float* __restrict__ x,
              const __bf16* __restrict__ W1s,
              const float* __restrict__ b1,
              const __bf16* __restrict__ W2s,
              const float* __restrict__ b2,
              const float* __restrict__ gates,
              float* __restrict__ out) {
    extern __shared__ __align__(16) unsigned char smem[];
    __bf16* sX = (__bf16*)smem;                          // ROWS x XSTR
    __bf16* sH = sX + ROWS * XSTR;                       // ROWS x CSTR
    float*  sG = (float*)(sH + ROWS * CSTR);             // ROWS x 4 gates

    const int tid  = threadIdx.x;
    const int wave = tid >> 5, lane = tid & 31;
    const int half = lane >> 4, n = lane & 15;
    const int rgrp = wave >> 3;          // row half: 0 -> rows 0-15, 1 -> 16-31
    const int wv8  = wave & 7;           // wave within row half
    const int rowBase = blockIdx.x * ROWS;
    const int rsub = rgrp * 16;

    // stage X tile (32 x 1024) fp32 -> bf16
    for (int idx = tid; idx < ROWS * EE; idx += 512) {
        int r = idx >> 10, c = idx & (EE - 1);
        sX[r * XSTR + c] = (__bf16)x[(size_t)(rowBase + r) * EE + c];
    }
    if (tid < ROWS * KEXP) sG[tid] = gates[(size_t)rowBase * KEXP + tid];
    __syncthreads();

    v8f outAcc[8];
#pragma unroll
    for (int t = 0; t < 8; ++t) outAcc[t] = (v8f){0, 0, 0, 0, 0, 0, 0, 0};

    for (int k = 0; k < KEXP; ++k) {
        const __bf16* W1k = W1s + (size_t)k * EE * HH;
        const __bf16* W2k = W2s + (size_t)k * HH * EE;
        __builtin_prefetch(W1k, 0, 1);   // global_prefetch_b8: warm L2 path

        v8f cAcc[8];                      // down-proj accum across both chunks
#pragma unroll
        for (int t = 0; t < 8; ++t) cAcc[t] = (v8f){0, 0, 0, 0, 0, 0, 0, 0};

        for (int c = 0; c < NCH; ++c) {
            __syncthreads();              // sH reuse: prior phase-B reads done

            // ---- phase A: 2048 H-cols, 256 per wave (16 tiles) ----
            for (int t = 0; t < 16; ++t) {
                v8f acc = (v8f){0, 0, 0, 0, 0, 0, 0, 0};
                for (int kk = 0; kk < EE / 32; ++kk) {       // 32 k-steps
                    v16bf a = afrag(&sX[(rsub + (lane & 15)) * XSTR + kk * 32],
                                    half);
                    v16bf b = bfrag_sw(
                        W1k, kk * (HH / 16) + c * (CH / 16) + wv8 * 16 + t,
                        lane);
                    acc = wmma_bf16(a, b, acc);
                }
                int hloc = wv8 * 256 + t * 16 + n;           // col in chunk
                float bias = b1[k * HH + c * CH + hloc];
#pragma unroll
                for (int v = 0; v < 8; ++v)
                    sH[(rsub + v + half * 8) * CSTR + hloc] =
                        (__bf16)geluf(acc[v] + bias);
            }
            __syncthreads();

            // ---- phase B: partial down-proj over this chunk (64 k-steps) --
            for (int kk = 0; kk < CH / 32; ++kk) {
                v16bf a = afrag(&sH[(rsub + (lane & 15)) * CSTR + kk * 32],
                                half);
#pragma unroll
                for (int t = 0; t < 8; ++t) {
                    v16bf b = bfrag_sw(
                        W2k, (c * (CH / 32) + kk) * (EE / 16) + wv8 * 8 + t,
                        lane);
                    cAcc[t] = wmma_bf16(a, b, cAcc[t]);
                }
            }
        }

        // ---- gate * (o + b2) accumulation ----
        float gk[8];
#pragma unroll
        for (int v = 0; v < 8; ++v)
            gk[v] = sG[(rsub + v + half * 8) * KEXP + k];
#pragma unroll
        for (int t = 0; t < 8; ++t) {
            float b2v = b2[k * EE + wv8 * 128 + t * 16 + n];
#pragma unroll
            for (int v = 0; v < 8; ++v)
                outAcc[t][v] += gk[v] * (cAcc[t][v] + b2v);
        }
    }

    // ---- write output (each element exactly once) ----
#pragma unroll
    for (int t = 0; t < 8; ++t) {
        int ecol = wv8 * 128 + t * 16 + n;
#pragma unroll
        for (int v = 0; v < 8; ++v)
            out[(size_t)(rowBase + rsub + v + half * 8) * EE + ecol] =
                outAcc[t][v];
    }
}

// ---------------------------------------------------------------------------
extern "C" void kernel_launch(void* const* d_in, const int* in_sizes, int n_in,
                              void* d_out, int out_size, void* d_ws, size_t ws_size,
                              hipStream_t stream) {
    const float* x     = (const float*)d_in[0];   // (B, M, E)
    const float* Wg1   = (const float*)d_in[1];   // (3E, E)
    const float* bg1   = (const float*)d_in[2];   // (E)
    const float* gamma = (const float*)d_in[3];   // (E)
    const float* beta  = (const float*)d_in[4];   // (E)
    const float* Wg2   = (const float*)d_in[5];   // (E, 12)
    const float* bg2   = (const float*)d_in[6];   // (12)
    const float* W1    = (const float*)d_in[7];   // (K, E, H)
    const float* b1    = (const float*)d_in[8];   // (K, H)
    const float* W2    = (const float*)d_in[9];   // (K, H, E)
    const float* b2    = (const float*)d_in[10];  // (K, E)
    float* out = (float*)d_out;                   // (B, M, E)

    // workspace layout (all 16B-aligned):
    //   gact (B*E f32) | gates (B*12 f32) | Wg1s | W1s | W2s  (bf16 swizzled)
    unsigned char* w = (unsigned char*)d_ws;
    float*  gact  = (float*)w;                     w += (size_t)BB * EE * 4;
    float*  gates = (float*)w;                     w += (size_t)BB * 12 * 4;
    __bf16* Wg1s  = (__bf16*)w;                    w += (size_t)3 * EE * EE * 2;
    __bf16* W1s   = (__bf16*)w;                    w += (size_t)KEXP * EE * HH * 2;
    __bf16* W2s   = (__bf16*)w;

    const size_t shm1 = (size_t)16 * GSTR3 * 2 + (size_t)16 * RSTR * 4;   // ~160 KB
    const size_t shm3 = (size_t)ROWS * XSTR * 2 + (size_t)ROWS * CSTR * 2
                      + (size_t)ROWS * KEXP * 4;                          // ~198 KB

    // Allow >64 KB dynamic LDS (CDNA5 WGP has 320 KB); no-op if already set.
    (void)hipFuncSetAttribute((const void*)gate1_kernel,
        hipFuncAttributeMaxDynamicSharedMemorySize, (int)shm1);
    (void)hipFuncSetAttribute((const void*)expert_kernel,
        hipFuncAttributeMaxDynamicSharedMemorySize, (int)shm3);

    // ---- weight swizzle passes (fp32 -> bf16, B-fragment tile order) ----
    {
        int thr = (3 * EE * EE) / 16;     // Wg1: K=3072, N=1024
        wswizzle_kernel<<<dim3((thr + 255) / 256, 1), 256, 0, stream>>>(
            Wg1, Wg1s, 3 * EE, EE);
        thr = (EE * HH) / 16;             // W1[k]: K=E, N=H  (4 experts)
        wswizzle_kernel<<<dim3((thr + 255) / 256, KEXP), 256, 0, stream>>>(
            W1, W1s, EE, HH);
        thr = (HH * EE) / 16;             // W2[k]: K=H, N=E  (4 experts)
        wswizzle_kernel<<<dim3((thr + 255) / 256, KEXP), 256, 0, stream>>>(
            W2, W2s, HH, EE);
    }

    gate1_kernel<<<BB / 16, 256, shm1, stream>>>(x, Wg1s, bg1, gamma, beta, gact);
    gate2_kernel<<<BB, 64, 0, stream>>>(gact, Wg2, bg2, gates);
    expert_kernel<<<RTOT / ROWS, 512, shm3, stream>>>(x, W1s, b1, W2s, b2, gates, out);
}
